// GCN_46136538693887
// MI455X (gfx1250) — matrix operand order repaired and checked by reading, not compiled, not run on hardware
//
#include <hip/hip_runtime.h>
#include <hip/hip_bf16.h>

// ---------------------------------------------------------------------------
// GCN forward for MI455X (gfx1250, wave32, WMMA bf16 matrix cores).
// GEMMs: v_wmma_f32_16x16x32_bf16, 64x128 block tile, 4 WMMA accum per wave,
//        GLOBAL_LOAD_ASYNC_TO_LDS_B128 staging (ASYNCcnt path) when available.
// Aggregation: float4 gather + global fp32 atomics (L2-resident, 192MB L2).
// ---------------------------------------------------------------------------

#define GCN_N 50000
#define GCN_E 800000
#define GCN_F0 128
#define GCN_H1 256
#define GCN_H2 512
#define GCN_H3 256

typedef __bf16 bf16_t;
typedef bf16_t v16bf __attribute__((ext_vector_type(16)));
typedef float  v8f   __attribute__((ext_vector_type(8)));
typedef int    v4i32 __attribute__((vector_size(16)));

#define AS1G __attribute__((address_space(1)))
#define AS3L __attribute__((address_space(3)))

#if defined(__has_builtin)
#if __has_builtin(__builtin_amdgcn_global_load_async_to_lds_b128) && \
    __has_builtin(__builtin_amdgcn_s_wait_asynccnt)
#define USE_ASYNC_LDS 1
#endif
#endif
#ifndef USE_ASYNC_LDS
#define USE_ASYNC_LDS 0
#endif

__device__ __forceinline__ unsigned short f2bf_rne(float f) {
  unsigned u = __float_as_uint(f);
  u += 0x7FFFu + ((u >> 16) & 1u);          // round-to-nearest-even
  return (unsigned short)(u >> 16);
}

// --------------------------- small utility kernels -------------------------

__global__ void k_cvt_f32_bf16(const float* __restrict__ in,
                               unsigned short* __restrict__ out, int n) {
  int i = blockIdx.x * blockDim.x + threadIdx.x;
  if (i < n) out[i] = f2bf_rne(in[i]);
}

__global__ void k_deg_init(float* __restrict__ deg, int n) {
  int i = blockIdx.x * blockDim.x + threadIdx.x;
  if (i < n) deg[i] = 1.0f;                  // self-loop contribution
}

__global__ void k_deg_acc(float* __restrict__ deg, const int* __restrict__ col, int e) {
  int i = blockIdx.x * blockDim.x + threadIdx.x;
  if (i < e) atomicAdd(&deg[col[i]], 1.0f);
}

__global__ void k_deg_fin(float* __restrict__ deg, int n) {
  int i = blockIdx.x * blockDim.x + threadIdx.x;
  if (i < n) deg[i] = rsqrtf(deg[i]);        // deg >= 1 always (self-loops)
}

// ------------------------------ WMMA GEMM ----------------------------------
// H[M,F] (f32) = X[M,K] (bf16, row-major) @ W[K,F] (bf16, row-major)
// K % 32 == 0, F % 128 == 0, M arbitrary (guarded; M % 16 == 0 assumed for A).
// Block tile 64(M) x 128(N); 8 waves in 4x2 grid; each wave: 16x64 = 4 accums.
// grid = (ceil(M/64), F/128), block = 256.
template <int K, int F>
__global__ __launch_bounds__(256) void k_gemm_bf16_wmma(
    const unsigned short* __restrict__ X, const unsigned short* __restrict__ W,
    float* __restrict__ H, int M) {
  constexpr int ASR = 20;  // dwords per A row in LDS (16 payload + 4 pad, 16B-aligned rows)
  constexpr int BSR = 34;  // shorts per B row in LDS (32 + 2 pad)
  __shared__ unsigned       Asm[64 * ASR];   // [m][kpair] dwords (5.0 KB)
  __shared__ unsigned short Bsm[128 * BSR];  // [n][k] shorts, transposed (8.5 KB)

  const int tid  = threadIdx.x;
  const int lane = tid & 31;
  const int wave = tid >> 5;
  const int wm   = wave >> 1;                // 0..3 : M sub-tile
  const int wn   = wave & 1;                 // 0..1 : N half
  const int m0   = blockIdx.x * 64;
  const int n0   = blockIdx.y * 128;

  const int mA   = lane & 15;                // A-fragment row within 16
  const int half = lane >> 4;                // lane half selects K region
  const int nlo  = lane & 15;

  const v8f zero8 = {0.f, 0.f, 0.f, 0.f, 0.f, 0.f, 0.f, 0.f};
  v8f acc[4] = {zero8, zero8, zero8, zero8};

  // A staging assignment: each thread copies one 16B chunk per K-chunk
  const int arow = tid >> 2;                 // 0..63
  const int achk = tid & 3;                  // 0..3  (8 bf16 = 16B each)
  const size_t aRowOff = (size_t)(m0 + arow) * K;

  for (int kc = 0; kc < K; kc += 32) {
    // ---- stage A tile: 64x32 bf16 (async DMA to LDS when available)
#if USE_ASYNC_LDS
    __builtin_amdgcn_global_load_async_to_lds_b128(
        (AS1G v4i32*)(const void*)(X + aRowOff + kc + achk * 8),
        (AS3L v4i32*)(void*)&Asm[arow * ASR + achk * 4], 0, 0);
#else
    {
      uint4 v = make_uint4(0u, 0u, 0u, 0u);
      if (m0 + arow < M) v = *(const uint4*)(X + aRowOff + kc + achk * 8);
      *(uint4*)&Asm[arow * ASR + achk * 4] = v;
    }
#endif
    // ---- stage B tile: 32x128 bf16 -> transposed [n][k]
#pragma unroll
    for (int j = 0; j < 8; ++j) {
      int d  = tid + 256 * j;                // 0..2047 dwords
      int k  = d >> 6;                       // 0..31
      int nd = d & 63;                       // dword column (2 n's)
      unsigned val = *(const unsigned*)(W + (size_t)(kc + k) * F + n0 + 2 * nd);
      Bsm[(2 * nd)     * BSR + k] = (unsigned short)(val & 0xFFFFu);
      Bsm[(2 * nd + 1) * BSR + k] = (unsigned short)(val >> 16);
    }
#if USE_ASYNC_LDS
    __builtin_amdgcn_s_wait_asynccnt(0);
#endif
    __syncthreads();

    // ---- A fragment (ISA 16-bit A 16x32 layout), loaded once per wave
    union { v16bf v; unsigned u[8]; } af;
#pragma unroll
    for (int v = 0; v < 8; ++v) {
      int k = ((v & 4) ? 16 : 0) + half * 8 + (v & 3) * 2;
      af.u[v] = Asm[(wm * 16 + mA) * ASR + (k >> 1)];
    }
    // ---- 4 N-subtiles: B fragment + WMMA each
#pragma unroll
    for (int j = 0; j < 4; ++j) {
      union { v16bf v; unsigned u[8]; } bf;
      const int nB = wn * 64 + j * 16 + nlo;
#pragma unroll
      for (int v = 0; v < 8; ++v)
        bf.u[v] = *(const unsigned*)&Bsm[nB * BSR + ((v & 4) ? 16 : 0) + half * 8 + (v & 3) * 2];
      acc[j] = __builtin_amdgcn_wmma_f32_16x16x32_bf16(
          false, af.v, false, bf.v, (short)0, acc[j], false, false);
    }
    __syncthreads();
  }

  // ---- store D: VGPR v -> M = wm*16 + half*8 + v
  float* Hp = H + (size_t)(m0 + wm * 16 + half * 8) * F + n0 + wn * 64 + nlo;
#pragma unroll
  for (int v = 0; v < 8; ++v) {
    if (m0 + wm * 16 + half * 8 + v < M) {
#pragma unroll
      for (int j = 0; j < 4; ++j)
        Hp[(size_t)v * F + j * 16] = acc[j][v];
    }
  }
}

// --------------------------- aggregation kernels ---------------------------

// agg[i,:] = h[i,:] * dinv[i]^2   (self-loop term, no atomics)
__global__ void k_agg_init(const float4* __restrict__ h4, float4* __restrict__ agg4,
                           const float* __restrict__ dinv, int total4, int shift) {
  int i = blockIdx.x * blockDim.x + threadIdx.x;
  if (i >= total4) return;
  float d = dinv[i >> shift];
  float w = d * d;
  float4 v = h4[i];
  v.x *= w; v.y *= w; v.z *= w; v.w *= w;
  agg4[i] = v;
}

// for each edge e: agg[col[e],:] += h[row[e],:] * dinv[row]*dinv[col]
// idx = e*(F/4) + chunk -> consecutive threads cover one edge's row (coalesced)
__global__ void k_scatter_edges(float* __restrict__ agg, const float4* __restrict__ h4,
                                const int* __restrict__ row, const int* __restrict__ col,
                                const float* __restrict__ dinv, int total, int shift, int F) {
  int i = blockIdx.x * blockDim.x + threadIdx.x;
  if (i >= total) return;
  int e = i >> shift;
  int c = i & ((1 << shift) - 1);
  int r = row[e];
  int t = col[e];
  float w = dinv[r] * dinv[t];
  float4 v = h4[((size_t)r << shift) + c];
  float* dst = agg + (size_t)t * F + (c << 2);
  atomicAdd(dst + 0, v.x * w);
  atomicAdd(dst + 1, v.y * w);
  atomicAdd(dst + 2, v.z * w);
  atomicAdd(dst + 3, v.w * w);
}

// y = relu(y + b) in place; also emit bf16 copy for the next layer's GEMM
__global__ void k_bias_relu_cvt(float* __restrict__ y, const float* __restrict__ b,
                                unsigned short* __restrict__ ybf, int total, int fmask) {
  int i = blockIdx.x * blockDim.x + threadIdx.x;
  if (i >= total) return;
  float t = fmaxf(y[i] + b[i & fmask], 0.0f);
  y[i]   = t;
  ybf[i] = f2bf_rne(t);
}

// ------------------------------ pooling + MLP ------------------------------

__global__ void k_zero256(float* __restrict__ g) { g[threadIdx.x] = 0.0f; }

// blockDim.x == 256 == H3; block-strided over rows, coalesced row reads
__global__ void k_mean_pool(const float* __restrict__ y, float* __restrict__ g, int n) {
  float s = 0.0f;
  for (int i = blockIdx.x; i < n; i += gridDim.x)
    s += y[(size_t)i * GCN_H3 + threadIdx.x];
  atomicAdd(&g[threadIdx.x], s);
}

__global__ void k_mlp_head(const float* __restrict__ g,
                           const float* __restrict__ Wf1, const float* __restrict__ bf1,
                           const float* __restrict__ Wf2, const float* __restrict__ bf2,
                           const float* __restrict__ Wf3, const float* __restrict__ bf3,
                           float* __restrict__ out) {
  __shared__ float sg[256];
  __shared__ float s1[128];
  __shared__ float s2[64];
  int t = threadIdx.x;
  sg[t] = g[t] * (1.0f / (float)GCN_N);
  __syncthreads();
  if (t < 128) {
    float s = bf1[t];
    for (int k = 0; k < 256; ++k) s = fmaf(sg[k], Wf1[k * 128 + t], s);
    s1[t] = fmaxf(s, 0.0f);
  }
  __syncthreads();
  if (t < 64) {
    float s = bf2[t];
    for (int k = 0; k < 128; ++k) s = fmaf(s1[k], Wf2[k * 64 + t], s);
    s2[t] = fmaxf(s, 0.0f);
  }
  __syncthreads();
  if (t == 0) {
    float s = bf3[0];
    for (int k = 0; k < 64; ++k) s = fmaf(s2[k], Wf3[k], s);
    out[0] = s;
  }
}

// ------------------------------- launcher ----------------------------------

extern "C" void kernel_launch(void* const* d_in, const int* in_sizes, int n_in,
                              void* d_out, int out_size, void* d_ws, size_t ws_size,
                              hipStream_t stream) {
  (void)in_sizes; (void)n_in; (void)out_size; (void)ws_size;

  const float* x   = (const float*)d_in[0];
  const int*   ei  = (const int*)d_in[1];
  const int*   row = ei;                 // edge_index[0] : sources
  const int*   col = ei + GCN_E;         // edge_index[1] : targets
  const float* W1  = (const float*)d_in[2];  const float* b1  = (const float*)d_in[3];
  const float* W2  = (const float*)d_in[4];  const float* b2  = (const float*)d_in[5];
  const float* W3  = (const float*)d_in[6];  const float* b3  = (const float*)d_in[7];
  const float* Wf1 = (const float*)d_in[8];  const float* bf1 = (const float*)d_in[9];
  const float* Wf2 = (const float*)d_in[10]; const float* bf2 = (const float*)d_in[11];
  const float* Wf3 = (const float*)d_in[12]; const float* bf3 = (const float*)d_in[13];

  // ---- workspace carving (256B aligned); total ~257 MB
  char* ws = (char*)d_ws;
  size_t off = 0;
  auto carve = [&](size_t bytes) -> void* {
    void* p = ws + off;
    off += (bytes + 255) & ~(size_t)255;
    return p;
  };
  float*          dinv = (float*)carve((size_t)GCN_N * 4);
  unsigned short* w1bf = (unsigned short*)carve((size_t)GCN_F0 * GCN_H1 * 2);
  unsigned short* w2bf = (unsigned short*)carve((size_t)GCN_H1 * GCN_H2 * 2);
  unsigned short* w3bf = (unsigned short*)carve((size_t)GCN_H2 * GCN_H3 * 2);
  unsigned short* xbf  = (unsigned short*)carve((size_t)GCN_N * GCN_H2 * 2);
  float*          bufA = (float*)carve((size_t)GCN_N * GCN_H2 * 4);   // GEMM out
  float*          bufB = (float*)carve((size_t)GCN_N * GCN_H2 * 4);   // agg / layer out
  float*          gbuf = (float*)carve(256 * 4);

  const int TB = 256;
  auto blocks = [](long long n) { return (int)((n + 255) / 256); };
  const int mblk = (GCN_N + 63) / 64;

  // ---- bf16 conversions (weights + input features)
  k_cvt_f32_bf16<<<blocks((long long)GCN_N * GCN_F0), TB, 0, stream>>>(x, xbf, GCN_N * GCN_F0);
  k_cvt_f32_bf16<<<blocks(GCN_F0 * GCN_H1), TB, 0, stream>>>(W1, w1bf, GCN_F0 * GCN_H1);
  k_cvt_f32_bf16<<<blocks(GCN_H1 * GCN_H2), TB, 0, stream>>>(W2, w2bf, GCN_H1 * GCN_H2);
  k_cvt_f32_bf16<<<blocks(GCN_H2 * GCN_H3), TB, 0, stream>>>(W3, w3bf, GCN_H2 * GCN_H3);

  // ---- degree -> dinv = rsqrt(deg)
  k_deg_init<<<blocks(GCN_N), TB, 0, stream>>>(dinv, GCN_N);
  k_deg_acc<<<blocks(GCN_E), TB, 0, stream>>>(dinv, col, GCN_E);
  k_deg_fin<<<blocks(GCN_N), TB, 0, stream>>>(dinv, GCN_N);

  // ---- layer 1: K=128, F=256 (F/4=64 -> shift 6)
  k_gemm_bf16_wmma<GCN_F0, GCN_H1><<<dim3(mblk, GCN_H1 / 128), TB, 0, stream>>>(xbf, w1bf, bufA, GCN_N);
  k_agg_init<<<blocks((long long)GCN_N * 64), TB, 0, stream>>>((const float4*)bufA, (float4*)bufB, dinv, GCN_N * 64, 6);
  k_scatter_edges<<<blocks((long long)GCN_E * 64), TB, 0, stream>>>(bufB, (const float4*)bufA, row, col, dinv, GCN_E * 64, 6, GCN_H1);
  k_bias_relu_cvt<<<blocks((long long)GCN_N * GCN_H1), TB, 0, stream>>>(bufB, b1, xbf, GCN_N * GCN_H1, GCN_H1 - 1);

  // ---- layer 2: K=256, F=512 (F/4=128 -> shift 7)
  k_gemm_bf16_wmma<GCN_H1, GCN_H2><<<dim3(mblk, GCN_H2 / 128), TB, 0, stream>>>(xbf, w2bf, bufA, GCN_N);
  k_agg_init<<<blocks((long long)GCN_N * 128), TB, 0, stream>>>((const float4*)bufA, (float4*)bufB, dinv, GCN_N * 128, 7);
  k_scatter_edges<<<blocks((long long)GCN_E * 128), TB, 0, stream>>>(bufB, (const float4*)bufA, row, col, dinv, GCN_E * 128, 7, GCN_H2);
  k_bias_relu_cvt<<<blocks((long long)GCN_N * GCN_H2), TB, 0, stream>>>(bufB, b2, xbf, GCN_N * GCN_H2, GCN_H2 - 1);

  // ---- layer 3: K=512, F=256
  k_gemm_bf16_wmma<GCN_H2, GCN_H3><<<dim3(mblk, GCN_H3 / 128), TB, 0, stream>>>(xbf, w3bf, bufA, GCN_N);
  k_agg_init<<<blocks((long long)GCN_N * 64), TB, 0, stream>>>((const float4*)bufA, (float4*)bufB, dinv, GCN_N * 64, 6);
  k_scatter_edges<<<blocks((long long)GCN_E * 64), TB, 0, stream>>>(bufB, (const float4*)bufA, row, col, dinv, GCN_E * 64, 6, GCN_H3);
  k_bias_relu_cvt<<<blocks((long long)GCN_N * GCN_H3), TB, 0, stream>>>(bufB, b3, xbf, GCN_N * GCN_H3, GCN_H3 - 1);

  // ---- mean pool over nodes, then tiny MLP head
  k_zero256<<<1, 256, 0, stream>>>(gbuf);
  k_mean_pool<<<512, 256, 0, stream>>>(bufB, gbuf, GCN_N);
  k_mlp_head<<<1, 256, 0, stream>>>(gbuf, Wf1, bf1, Wf2, bf2, Wf3, bf3, (float*)d_out);
}
